// MultiHeadSelfAttention_69020124447146
// MI455X (gfx1250) — matrix operand order repaired and checked by reading, not compiled, not run on hardware
//
#include <hip/hip_runtime.h>
#include <stdint.h>

// MHA forward for B=4, S=2048, D=1024, H=16, HD=64 on gfx1250 (wave32, WMMA).
// Pipeline: fp32->bf16 convert, 3 WMMA GEMMs (QKV proj, V stored transposed),
// flash-style WMMA attention with TDM (tensor_load_to_lds) double-buffered
// K/V staging, final WMMA GEMM (output proj, fp32 out).

typedef __attribute__((ext_vector_type(16))) __bf16 v16bf;
typedef __attribute__((ext_vector_type(8)))  float  v8f;
typedef __attribute__((ext_vector_type(4)))  unsigned int v4u;
typedef __attribute__((ext_vector_type(8)))  int v8i_t;
typedef __attribute__((ext_vector_type(4)))  int v4i_t;

#define S_LEN  2048
#define DMODEL 1024
#define NHEAD  16
#define HDIM   64
#define BATCH  4
#define MTOT   (BATCH * S_LEN)   // 8192
#define KC     64                // keys staged per TDM chunk

union Frag {
  v16bf v;
  uint4 q[2];
};

__device__ __forceinline__ uint16_t f2bf(float f) {
  uint32_t u = __builtin_bit_cast(uint32_t, f);
  u += 0x7FFFu + ((u >> 16) & 1u);   // round-to-nearest-even
  return (uint16_t)(u >> 16);
}

__device__ __forceinline__ v8f wmma_bf16(v16bf a, v16bf b, v8f c) {
  // D = A(16x32 bf16) x B(32x16 bf16) + C(16x16 f32)
  return __builtin_amdgcn_wmma_f32_16x16x32_bf16(
      false, a, false, b, (short)0, c, false, false);
}

// TDM: DMA a 2D bf16 tile (tl0 contiguous elems x tl1 rows) from a row-major
// tensor (row stride stride0 elems) in global memory into LDS at lds_off.
// D# layout per CDNA5 ISA ch.8: group0 = count/lds/global/type, group1 =
// data_size + tensor dims + tile dims + dim0 stride. Groups 2/3 unused (2D).
// This toolchain's builtin takes 6 args (g0, g1, g2, g3, g4, cpol).
__device__ __forceinline__ void tdm_load_2d(uint64_t gaddr, uint32_t lds_off,
                                            uint32_t td0, uint32_t td1,
                                            uint32_t tl0, uint32_t tl1,
                                            uint64_t stride0) {
  v4u g0;
  g0[0] = 1u;                                   // count=1, user descriptor
  g0[1] = lds_off;                              // lds_addr (bytes)
  g0[2] = (uint32_t)gaddr;                      // global_addr[31:0]
  g0[3] = ((uint32_t)(gaddr >> 32) & 0x01FFFFFFu) | 0x80000000u; // type=2
  v8i_t g1;
  g1[0] = 0x10000;                              // workgroup_mask=0, data_size=2B
  g1[1] = (int)((td0 & 0xFFFFu) << 16);                         // tensor_dim0 lo
  g1[2] = (int)(((td0 >> 16) & 0xFFFFu) | ((td1 & 0xFFFFu) << 16));
  g1[3] = (int)(((td1 >> 16) & 0xFFFFu) | ((tl0 & 0xFFFFu) << 16));
  g1[4] = (int)(tl1 & 0xFFFFu);                                 // tile_dim1
  g1[5] = (int)(uint32_t)stride0;                               // dim0 stride lo
  g1[6] = (int)((stride0 >> 32) & 0xFFFFu);                     // stride hi
  g1[7] = 0;
  v4i_t z4 = {0, 0, 0, 0};
  v8i_t z8 = {0, 0, 0, 0, 0, 0, 0, 0};
  __builtin_amdgcn_tensor_load_to_lds(g0, g1, z4, z4, z8, 0);
}

__global__ void cvt_f32_bf16(const float* __restrict__ s,
                             uint16_t* __restrict__ d, int n) {
  int i = blockIdx.x * blockDim.x + threadIdx.x;
  if (i < n) d[i] = f2bf(s[i]);
}

// C[M,N] = A[M,K] * W[N,K]^T + bias  (Linear semantics), K = N = DMODEL.
// mode 0: fp32 store to [M, DMODEL] (final output)
// mode 1: bf16 store scattered to [b,h,s,hd]      (Q with scale, K)
// mode 2: bf16 store scattered to [b,h,hd,s]      (V transposed)
__global__ __launch_bounds__(256)
void gemm_bf16(const uint16_t* __restrict__ A, const uint16_t* __restrict__ W,
               const float* __restrict__ bias, void* __restrict__ dst,
               int mode, float scale) {
  const int lane = threadIdx.x & 31;
  const int wave = threadIdx.x >> 5;
  const int r    = lane & 15;
  const int hi   = lane >> 4;
  const int mbase = blockIdx.y * 128 + (wave >> 2) * 64;
  const int nbase = blockIdx.x * 256 + (wave & 3) * 64;

  v8f acc[4][4] = {};

  for (int kk = 0; kk < DMODEL; kk += 32) {
    Frag a[4];
#pragma unroll
    for (int mt = 0; mt < 4; ++mt) {
      const uint16_t* ap = A + (size_t)(mbase + mt * 16 + r) * DMODEL + kk;
      a[mt].q[0] = *(const uint4*)(ap + hi * 16);       // K = hi*16 .. +7
      a[mt].q[1] = *(const uint4*)(ap + hi * 16 + 8);   // K = hi*16+8 .. +15
    }
#pragma unroll
    for (int nt = 0; nt < 4; ++nt) {
      Frag b;
      const uint16_t* wp =
          W + (size_t)(nbase + nt * 16 + r) * DMODEL + kk + hi * 16;
      b.q[0] = *(const uint4*)(wp);                     // K = hi*16 .. +7
      b.q[1] = *(const uint4*)(wp + 8);                 // K = hi*16+8 .. +15
#pragma unroll
      for (int mt = 0; mt < 4; ++mt)
        acc[mt][nt] = wmma_bf16(a[mt].v, b.v, acc[mt][nt]);
    }
  }

#pragma unroll
  for (int mt = 0; mt < 4; ++mt) {
#pragma unroll
    for (int nt = 0; nt < 4; ++nt) {
      const int col = nbase + nt * 16 + r;
      const float bv = bias[col];
#pragma unroll
      for (int j = 0; j < 8; ++j) {
        const int row = mbase + mt * 16 + j + 8 * hi;   // C layout: M=j+8*hi
        float val = acc[mt][nt][j] + bv;
        if (mode == 0) {
          ((float*)dst)[(size_t)row * DMODEL + col] = val;
        } else {
          const int b  = row >> 11, s  = row & (S_LEN - 1);
          const int h  = col >> 6,  hd = col & (HDIM - 1);
          uint16_t* o = (uint16_t*)dst;
          if (mode == 1)
            o[((size_t)(b * NHEAD + h) * S_LEN + s) * HDIM + hd] =
                f2bf(val * scale);
          else
            o[((size_t)(b * NHEAD + h) * HDIM + hd) * S_LEN + s] = f2bf(val);
        }
      }
    }
  }
}

// Flash-style attention. Block = 8 waves sharing one (b,h); each wave owns 16
// query rows (128 q-rows/block). K (64x64) and V^T (64x64) tiles are staged
// into LDS by the Tensor Data Mover, double-buffered so the DMA for chunk c+1
// overlaps WMMA compute on chunk c. Output ctx bf16 [b,s,d].
__global__ __launch_bounds__(256)
void attention_bf16(const uint16_t* __restrict__ Q,
                    const uint16_t* __restrict__ K,
                    const uint16_t* __restrict__ Vt,
                    uint16_t* __restrict__ ctx) {
  __shared__ __align__(16) uint16_t Kls[2][KC * HDIM];   // 2 x 8 KB, [key][hd]
  __shared__ __align__(16) uint16_t Vls[2][HDIM * KC];   // 2 x 8 KB, [hd][key]
  __shared__ __align__(16) uint16_t Pls[8][16 * 32];     // 1 KB P-tile per wave

  const int lane = threadIdx.x & 31;
  const int wave = threadIdx.x >> 5;
  const int r    = lane & 15;
  const int hi   = lane >> 4;
  const int bh   = blockIdx.x >> 4;                 // 0..63
  const int qb   = blockIdx.x & 15;                 // 128-row q block
  const int b    = bh >> 4, h = bh & 15;
  const int qbase = qb * 128 + wave * 16;

  const uint16_t* Qb = Q  + (size_t)bh * S_LEN * HDIM;
  const uint16_t* Kb = K  + (size_t)bh * S_LEN * HDIM;
  const uint16_t* Vb = Vt + (size_t)bh * HDIM * S_LEN;

  // Q fragments (A-layout) for K-dim (HD=64) = 2 steps; Q pre-scaled by 1/8.
  Frag qf[2];
#pragma unroll
  for (int ks = 0; ks < 2; ++ks) {
    const uint16_t* qp = Qb + (size_t)(qbase + r) * HDIM + ks * 32;
    qf[ks].q[0] = *(const uint4*)(qp + hi * 16);
    qf[ks].q[1] = *(const uint4*)(qp + hi * 16 + 8);
  }

  v8f acc[4] = {};
  float mrow[8], lrow[8];
#pragma unroll
  for (int j = 0; j < 8; ++j) { mrow[j] = -__builtin_inff(); lrow[j] = 0.f; }

  // Prologue: TDM-load chunk 0 (one wave issues; EXEC is ignored by TDM).
  if (wave == 0) {
    tdm_load_2d((uint64_t)(uintptr_t)Kb, (uint32_t)(uintptr_t)&Kls[0][0],
                HDIM, S_LEN, HDIM, KC, HDIM);
    tdm_load_2d((uint64_t)(uintptr_t)Vb, (uint32_t)(uintptr_t)&Vls[0][0],
                S_LEN, HDIM, KC, HDIM, S_LEN);
  }

  for (int c = 0; c < S_LEN / KC; ++c) {
    const int buf = c & 1;
    __builtin_amdgcn_s_wait_tensorcnt(0);  // chunk c landed in LDS (wave 0)
    __syncthreads();                       // publish c; retire reads of c-1
    if (wave == 0 && c + 1 < S_LEN / KC) { // prefetch chunk c+1 into other buf
      const size_t koff = (size_t)(c + 1) * KC;
      tdm_load_2d((uint64_t)(uintptr_t)(Kb + koff * HDIM),
                  (uint32_t)(uintptr_t)&Kls[buf ^ 1][0],
                  HDIM, S_LEN, HDIM, KC, HDIM);
      tdm_load_2d((uint64_t)(uintptr_t)(Vb + koff),
                  (uint32_t)(uintptr_t)&Vls[buf ^ 1][0],
                  S_LEN, HDIM, KC, HDIM, S_LEN);
    }

#pragma unroll
    for (int kc2 = 0; kc2 < 2; ++kc2) {    // two 32-key sub-steps per chunk
      // S = Q * K^T: two 16x16 score tiles, K-dim 64 = 2 wmma each.
      Frag kf[2][2];
#pragma unroll
      for (int t = 0; t < 2; ++t)
#pragma unroll
        for (int ks = 0; ks < 2; ++ks) {
          const uint16_t* kp =
              &Kls[buf][(kc2 * 32 + t * 16 + r) * HDIM + ks * 32 + hi * 16];
          kf[t][ks].q[0] = *(const uint4*)(kp);
          kf[t][ks].q[1] = *(const uint4*)(kp + 8);
        }
      v8f s0 = {}, s1 = {};
      s0 = wmma_bf16(qf[0].v, kf[0][0].v, s0);
      s0 = wmma_bf16(qf[1].v, kf[0][1].v, s0);
      s1 = wmma_bf16(qf[0].v, kf[1][0].v, s1);
      s1 = wmma_bf16(qf[1].v, kf[1][1].v, s1);

      // Online softmax over the 32 new columns; rows live in 16-lane halves.
#pragma unroll
      for (int j = 0; j < 8; ++j) {
        float v = fmaxf(s0[j], s1[j]);
#pragma unroll
        for (int off = 8; off > 0; off >>= 1)
          v = fmaxf(v, __shfl_xor(v, off, 16));
        const float mn = fmaxf(mrow[j], v);
        const float al = __expf(mrow[j] - mn);
        const float p0 = __expf(s0[j] - mn);
        const float p1 = __expf(s1[j] - mn);
        float rs = p0 + p1;
#pragma unroll
        for (int off = 8; off > 0; off >>= 1)
          rs += __shfl_xor(rs, off, 16);
        lrow[j] = lrow[j] * al + rs;
        mrow[j] = mn;
#pragma unroll
        for (int nt = 0; nt < 4; ++nt) acc[nt][j] *= al;
        const int prow = j + 8 * hi;        // C layout -> row-major LDS
        Pls[wave][prow * 32 + r]      = f2bf(p0);
        Pls[wave][prow * 32 + 16 + r] = f2bf(p1);
      }

      // Re-read P in A-fragment layout (wave-private LDS round trip).
      Frag pf;
      const uint16_t* pp = &Pls[wave][r * 32];
      pf.q[0] = *(const uint4*)(pp + hi * 16);
      pf.q[1] = *(const uint4*)(pp + hi * 16 + 8);

      // ctx += P(16x32) * V(32x64): 4 hd tiles; V^T rows contiguous in keys.
#pragma unroll
      for (int nt = 0; nt < 4; ++nt) {
        Frag vf;
        const uint16_t* vp =
            &Vls[buf][(nt * 16 + r) * KC + kc2 * 32 + hi * 16];
        vf.q[0] = *(const uint4*)(vp);
        vf.q[1] = *(const uint4*)(vp + 8);
        acc[nt] = wmma_bf16(pf.v, vf.v, acc[nt]);
      }
    }
  }

#pragma unroll
  for (int j = 0; j < 8; ++j) {
    const float inv = 1.0f / lrow[j];
    const int srow = qbase + j + 8 * hi;
    uint16_t* op = ctx + (size_t)(b * S_LEN + srow) * DMODEL + h * HDIM;
#pragma unroll
    for (int nt = 0; nt < 4; ++nt)
      op[nt * 16 + r] = f2bf(acc[nt][j] * inv);
  }
}

extern "C" void kernel_launch(void* const* d_in, const int* in_sizes, int n_in,
                              void* d_out, int out_size, void* d_ws, size_t ws_size,
                              hipStream_t stream) {
  (void)in_sizes; (void)n_in; (void)out_size; (void)ws_size;
  const float* x  = (const float*)d_in[0];
  const float* wq = (const float*)d_in[1];
  const float* bq = (const float*)d_in[2];
  const float* wk = (const float*)d_in[3];
  const float* bk = (const float*)d_in[4];
  const float* wv = (const float*)d_in[5];
  const float* bv = (const float*)d_in[6];
  const float* wo = (const float*)d_in[7];
  const float* bo = (const float*)d_in[8];

  uint8_t* ws = (uint8_t*)d_ws;
  const size_t MB = 1ull << 20;
  uint16_t* Xbf = (uint16_t*)(ws +  0 * MB);  // 16 MB
  uint16_t* Wqb = (uint16_t*)(ws + 16 * MB);  //  2 MB each
  uint16_t* Wkb = (uint16_t*)(ws + 18 * MB);
  uint16_t* Wvb = (uint16_t*)(ws + 20 * MB);
  uint16_t* Wob = (uint16_t*)(ws + 22 * MB);
  uint16_t* Qbf = (uint16_t*)(ws + 24 * MB);  // 16 MB [b,h,s,hd], pre-scaled
  uint16_t* Kbf = (uint16_t*)(ws + 40 * MB);  // 16 MB [b,h,s,hd]
  uint16_t* Vtb = (uint16_t*)(ws + 56 * MB);  // 16 MB [b,h,hd,s]
  uint16_t* Ctx = (uint16_t*)(ws + 72 * MB);  // 16 MB [b,s,d]

  const int nx = MTOT * DMODEL;     // 8388608
  const int nw = DMODEL * DMODEL;   // 1048576
  cvt_f32_bf16<<<nx / 256, 256, 0, stream>>>(x,  Xbf, nx);
  cvt_f32_bf16<<<nw / 256, 256, 0, stream>>>(wq, Wqb, nw);
  cvt_f32_bf16<<<nw / 256, 256, 0, stream>>>(wk, Wkb, nw);
  cvt_f32_bf16<<<nw / 256, 256, 0, stream>>>(wv, Wvb, nw);
  cvt_f32_bf16<<<nw / 256, 256, 0, stream>>>(wo, Wob, nw);

  dim3 ggrid(DMODEL / 256, MTOT / 128);  // (4, 64)
  gemm_bf16<<<ggrid, 256, 0, stream>>>(Xbf, Wqb, bq, Qbf, 1, 0.125f); // Q/sqrt(64)
  gemm_bf16<<<ggrid, 256, 0, stream>>>(Xbf, Wkb, bk, Kbf, 1, 1.0f);
  gemm_bf16<<<ggrid, 256, 0, stream>>>(Xbf, Wvb, bv, Vtb, 2, 1.0f);   // V transposed

  // 64 (b,h) pairs x 16 q-blocks of 128 rows = 1024 blocks.
  attention_bf16<<<BATCH * NHEAD * (S_LEN / 128), 256, 0, stream>>>(
      Qbf, Kbf, Vtb, Ctx);

  gemm_bf16<<<ggrid, 256, 0, stream>>>(Ctx, Wob, bo, d_out, 0, 1.0f);
}
